// CustomSwinTransformer_52621939310977
// MI455X (gfx1250) — matrix operand order repaired and checked by reading, not compiled, not run on hardware
//
#include <hip/hip_runtime.h>
#include <math.h>

// ---------------------------------------------------------------------------
// Swin-T (three-modality) forward for MI455X (gfx1250), wave32 + WMMA bf16.
// Dense GEMMs: v_wmma_f32_16x16x32_bf16, 32x64 tile per wave (8 WMMA/k-step).
// Attention q/k/v staged to LDS via async global->LDS (ASYNCcnt) path.
//
// d_in layout assumption:
//   [0]=data_DCE (32*3*224*224 f32), [1]=data_T2, [2]=data_T1,
//   [3]=csv (32*107 f32),
//   [4..] params leaves in JAX pytree (sorted-key) order:
//     fc_b, fc_w, norm_b, norm_g, patch_b, patch_w,
//     stages[0..3]: blocks[*]: (fc1_b, fc1_w, fc2_b, fc2_w, ln1_b, ln1_g,
//                               ln2_b, ln2_g, proj_b, proj_w, qkv_b, qkv_w, rpb)
//                   then merge: (ln_b, ln_g, red_w)   [stages 0..2 only]
// Workspace requirement: ~430 MB (carved below).
// ---------------------------------------------------------------------------

typedef __bf16 bf16;
typedef __attribute__((ext_vector_type(16))) __bf16 bf16x16;
typedef __attribute__((ext_vector_type(8)))  __bf16 bf16x8;
typedef __attribute__((ext_vector_type(8)))  float  v8f;

// --------------------------- utility kernels -------------------------------

__global__ void k_convert_pad(const float* __restrict__ src, bf16* __restrict__ dst,
                              int N, int K, int Kpad) {
  long long i = (long long)blockIdx.x * 256 + threadIdx.x;
  long long tot = (long long)N * Kpad;
  if (i >= tot) return;
  int k = (int)(i % Kpad);
  long long n = i / Kpad;
  dst[i] = (k < K) ? (bf16)src[n * K + k] : (bf16)0.f;
}

__global__ void k_im2col(const float* __restrict__ img, bf16* __restrict__ out, int B) {
  // img: (B,3,224,224), out: (B*56*56, 64) zero-padded past col 48
  long long i = (long long)blockIdx.x * 256 + threadIdx.x;
  long long tot = (long long)B * 3136 * 64;
  if (i >= tot) return;
  int col = (int)(i & 63);
  long long r = i >> 6;
  int pw = (int)(r % 56);
  int ph = (int)((r / 56) % 56);
  int b  = (int)(r / 3136);
  float v = 0.f;
  if (col < 48) {
    int ci = col >> 4, dh = (col >> 2) & 3, dw = col & 3;
    v = img[(((long long)(b * 3 + ci) * 224) + ph * 4 + dh) * 224 + pw * 4 + dw];
  }
  out[i] = (bf16)v;
}

__global__ void k_add(float* __restrict__ x, const float* __restrict__ y, long long n) {
  long long i = (long long)blockIdx.x * 256 + threadIdx.x;
  if (i < n) x[i] += y[i];
}

// --------------------------- WMMA GEMM -------------------------------------
// C[M,N] = A[M,K](bf16,row-major,lda) * W[N,K](bf16,row-major,ldw)^T + bias
// One wave computes a 32x(16*NF) tile: 2 A-fragments reused across NF
// B-fragments -> 2*NF WMMAs per 32-wide k-step. Fragment k order: contiguous
// (lane half h holds k = 16h..16h+15; A 16x32; B 32x16 = one 32B row chunk per lane; f32 C/D).

__device__ __forceinline__ float gelu_erf(float v) {
  return 0.5f * v * (1.f + erff(v * 0.70710678118654752f));
}

template <int NF>
__device__ __forceinline__ void gemm_wave(const bf16* __restrict__ A, int lda,
                                          const bf16* __restrict__ W, int ldw,
                                          const float* __restrict__ bias,
                                          void* __restrict__ out, int outBf16,
                                          int gelu, int N, int K,
                                          int mbase, int ntile0, int lane) {
  const bf16* pa0 = A + (long long)(mbase + (lane & 15)) * lda + ((lane >> 4) * 16);
  const bf16* pa1 = pa0 + (long long)16 * lda;
  const bf16* pb[NF];
#pragma unroll
  for (int j = 0; j < NF; ++j)
    pb[j] = W + (long long)((ntile0 + j) * 16 + (lane & 15)) * ldw + ((lane >> 4) * 16);

  v8f accL[NF] = {};
  v8f accH[NF] = {};
  for (int k0 = 0; k0 < K; k0 += 32) {
    // stream-prefetch A 512B ahead (speculative; OOB silently dropped)
    __builtin_prefetch(pa0 + k0 + 256, 0, 3);
    __builtin_prefetch(pa1 + k0 + 256, 0, 3);
    bf16x8 lo0 = *(const bf16x8*)(pa0 + k0);
    bf16x8 hi0 = *(const bf16x8*)(pa0 + k0 + 8);
    bf16x8 lo1 = *(const bf16x8*)(pa1 + k0);
    bf16x8 hi1 = *(const bf16x8*)(pa1 + k0 + 8);
    bf16x16 a0 = __builtin_shufflevector(lo0, hi0, 0,1,2,3,4,5,6,7,8,9,10,11,12,13,14,15);
    bf16x16 a1 = __builtin_shufflevector(lo1, hi1, 0,1,2,3,4,5,6,7,8,9,10,11,12,13,14,15);
#pragma unroll
    for (int j = 0; j < NF; ++j) {
      bf16x16 bfr = *(const bf16x16*)(pb[j] + k0);
      accL[j] = __builtin_amdgcn_wmma_f32_16x16x32_bf16(false, a0, false, bfr,
                                                        (short)0, accL[j], false, false);
      accH[j] = __builtin_amdgcn_wmma_f32_16x16x32_bf16(false, a1, false, bfr,
                                                        (short)0, accH[j], false, false);
    }
  }

  int mrow0 = mbase + ((lane >> 4) * 8);  // C/D: VGPR r -> row mrow0+r (lo half)
#pragma unroll
  for (int j = 0; j < NF; ++j) {
    int colN = (ntile0 + j) * 16 + (lane & 15);
    float bv = bias ? bias[colN] : 0.f;
#pragma unroll
    for (int r = 0; r < 8; ++r) {
      float v0 = accL[j][r] + bv;
      float v1 = accH[j][r] + bv;
      if (gelu) { v0 = gelu_erf(v0); v1 = gelu_erf(v1); }
      long long o0 = (long long)(mrow0 + r) * N + colN;
      long long o1 = (long long)(mrow0 + 16 + r) * N + colN;
      if (outBf16) { ((bf16*)out)[o0] = (bf16)v0; ((bf16*)out)[o1] = (bf16)v1; }
      else         { ((float*)out)[o0] = v0;      ((float*)out)[o1] = v1; }
    }
  }
}

__global__ __launch_bounds__(128)
void k_gemm(const bf16* __restrict__ A, int lda,
            const bf16* __restrict__ W, int ldw,
            const float* __restrict__ bias,
            void* __restrict__ out, int outBf16, int gelu,
            int M, int N, int K) {
  int lane = threadIdx.x;                   // 0..31
  int mt32 = blockIdx.x * 4 + threadIdx.y;  // 32-row tile index
  if (mt32 * 32 >= M) return;               // wave-uniform (M % 32 == 0)
  int ntile0 = blockIdx.y * 4;              // first 16-col tile
  int nf = N / 16 - ntile0;                 // N % 16 == 0
  int mbase = mt32 * 32;
  if (nf >= 4)      gemm_wave<4>(A, lda, W, ldw, bias, out, outBf16, gelu, N, K, mbase, ntile0, lane);
  else if (nf == 3) gemm_wave<3>(A, lda, W, ldw, bias, out, outBf16, gelu, N, K, mbase, ntile0, lane);
  else if (nf == 2) gemm_wave<2>(A, lda, W, ldw, bias, out, outBf16, gelu, N, K, mbase, ntile0, lane);
  else              gemm_wave<1>(A, lda, W, ldw, bias, out, outBf16, gelu, N, K, mbase, ntile0, lane);
}

// --------------------------- LayerNorm (one wave / row) --------------------
__global__ void k_layernorm(const float* __restrict__ x, const float* __restrict__ g,
                            const float* __restrict__ b, void* __restrict__ out,
                            int outBf16, int C) {
  long long row = blockIdx.x;
  int lane = threadIdx.x;
  const float* xr = x + row * C;
  float s = 0.f, s2 = 0.f;
  for (int c = lane; c < C; c += 32) { float v = xr[c]; s += v; s2 += v * v; }
  for (int off = 16; off; off >>= 1) {
    s  += __shfl_xor(s,  off, 32);
    s2 += __shfl_xor(s2, off, 32);
  }
  float mu  = s / C;
  float var = s2 / C - mu * mu;
  float inv = rsqrtf(var + 1e-5f);
  for (int c = lane; c < C; c += 32) {
    float v = (xr[c] - mu) * inv * g[c] + b[c];
    if (outBf16) ((bf16*)out)[row * C + c] = (bf16)v;
    else         ((float*)out)[row * C + c] = v;
  }
}

// ------------------- window partition (with shift) -------------------------
// xn: (B,H,H,C) bf16 -> win: (B*nW, 64, C) bf16, rows 49..63 zero.
__global__ void k_window_gather(const bf16* __restrict__ xn, bf16* __restrict__ win,
                                int B, int H, int C, int shift) {
  int nWr = H / 7;
  long long tot = (long long)B * nWr * nWr * 64 * C;
  long long i = (long long)blockIdx.x * 256 + threadIdx.x;
  if (i >= tot) return;
  int c = (int)(i % C);
  long long r = i / C;
  int t = (int)(r & 63);
  long long wq = r >> 6;
  int wx = (int)(wq % nWr); wq /= nWr;
  int wy = (int)(wq % nWr);
  int b  = (int)(wq / nWr);
  bf16 v = (bf16)0.f;
  if (t < 49) {
    int ty = t / 7, tx = t % 7;
    int hh = wy * 7 + ty + shift; if (hh >= H) hh -= H;  // roll(-shift)
    int ww = wx * 7 + tx + shift; if (ww >= H) ww -= H;
    v = xn[(((long long)b * H + hh) * H + ww) * C + c];
  }
  win[i] = v;
}

// ------------------- fused window attention (per window,head) --------------
// qkv: (B*nW*64, 3C) f32; att out: (B*nW*64, C) bf16 (pad rows zeroed). d=32.
// q/k/v staged into LDS with CDNA5 async global->LDS ops (ASYNCcnt).
__global__ __launch_bounds__(256)
void k_window_attn(const float* __restrict__ qkv, const float* __restrict__ rpb,
                   bf16* __restrict__ att, int C, int heads, int H, int shift,
                   int nWimg) {
  __shared__ float sq[49 * 32], sk[49 * 32], sv[49 * 32], sp[49 * 49];
  int blk  = blockIdx.x;
  int head = blk % heads;
  int win  = blk / heads;
  int wloc = win % nWimg;
  int nWr  = H / 7;
  int wy = wloc / nWr, wx = wloc % nWr;
  int tid = threadIdx.x;
  int threeC = 3 * C;
  long long base = (long long)win * 64;
  int qo = head * 32, ko = C + head * 32, vo = 2 * C + head * 32;

  // Async copy q/k/v tiles into LDS. LDS address is taken from the VDST VGPR; the
  // generic shared-pointer's low 32 bits are the LDS offset (aperture rule).
  for (int idx = tid; idx < 49 * 32; idx += 256) {
    int t = idx >> 5, k = idx & 31;
    const float* row = qkv + (base + t) * threeC;
    unsigned long long aq = (unsigned long long)(row + qo + k);
    unsigned long long ak = (unsigned long long)(row + ko + k);
    unsigned long long av = (unsigned long long)(row + vo + k);
    unsigned dq = (unsigned)(unsigned long long)(void*)&sq[idx];
    unsigned dk = (unsigned)(unsigned long long)(void*)&sk[idx];
    unsigned dv = (unsigned)(unsigned long long)(void*)&sv[idx];
    asm volatile("global_load_async_to_lds_b32 %0, %1, off" :: "v"(dq), "v"(aq) : "memory");
    asm volatile("global_load_async_to_lds_b32 %0, %1, off" :: "v"(dk), "v"(ak) : "memory");
    asm volatile("global_load_async_to_lds_b32 %0, %1, off" :: "v"(dv), "v"(av) : "memory");
  }
  asm volatile("s_wait_asynccnt 0x0" ::: "memory");
  __syncthreads();

  const float scale = 0.17677669529663687f;  // 32^-0.5
  for (int idx = tid; idx < 49 * 49; idx += 256) {
    int i = idx / 49, j = idx % 49;
    float d = 0.f;
#pragma unroll 8
    for (int k = 0; k < 32; ++k) d += sq[i * 32 + k] * sk[j * 32 + k];
    int ih = i / 7, iw = i % 7, jh = j / 7, jw = j % 7;
    int ridx = (ih - jh + 6) * 13 + (iw - jw + 6);
    float s = d * scale + rpb[ridx * heads + head];
    if (shift) {  // 9-region shifted-window mask, computed analytically
      int hi = wy * 7 + ih, wi = wx * 7 + iw;
      int hj = wy * 7 + jh, wj = wx * 7 + jw;
      int rHi = hi < H - 7 ? 0 : (hi < H - shift ? 1 : 2);
      int rWi = wi < H - 7 ? 0 : (wi < H - shift ? 1 : 2);
      int rHj = hj < H - 7 ? 0 : (hj < H - shift ? 1 : 2);
      int rWj = wj < H - 7 ? 0 : (wj < H - shift ? 1 : 2);
      if ((rHi * 3 + rWi) != (rHj * 3 + rWj)) s -= 100.f;
    }
    sp[idx] = s;
  }
  __syncthreads();

  if (tid < 49) {  // softmax, one thread per row (tiny)
    float mx = -1e30f;
    for (int j = 0; j < 49; ++j) mx = fmaxf(mx, sp[tid * 49 + j]);
    float sum = 0.f;
    for (int j = 0; j < 49; ++j) { float e = __expf(sp[tid * 49 + j] - mx); sp[tid * 49 + j] = e; sum += e; }
    float inv = 1.f / sum;
    for (int j = 0; j < 49; ++j) sp[tid * 49 + j] *= inv;
  }
  __syncthreads();

  for (int idx = tid; idx < 64 * 32; idx += 256) {
    int t = idx >> 5, k = idx & 31;
    float o = 0.f;
    if (t < 49)
      for (int j = 0; j < 49; ++j) o += sp[t * 49 + j] * sv[j * 32 + k];
    att[(base + t) * C + head * 32 + k] = (bf16)o;  // pad rows -> 0
  }
}

// ------------------- window reverse (+shift) with residual add -------------
__global__ void k_scatter_add(const float* __restrict__ proj, float* __restrict__ x,
                              int B, int H, int C, int shift) {
  int nWr = H / 7;
  long long tot = (long long)B * H * H * C;
  long long i = (long long)blockIdx.x * 256 + threadIdx.x;
  if (i >= tot) return;
  int c = (int)(i % C);
  long long r = i / C;
  int w = (int)(r % H);
  int h = (int)((r / H) % H);
  int b = (int)(r / ((long long)H * H));
  int hs = h - shift; if (hs < 0) hs += H;  // roll(+shift) inverse
  int ws_ = w - shift; if (ws_ < 0) ws_ += H;
  long long row = ((long long)b * nWr * nWr + (hs / 7) * nWr + (ws_ / 7)) * 64
                + (hs % 7) * 7 + (ws_ % 7);
  x[i] += proj[row * C + c];
}

// ------------------- patch-merge gather (2x2 -> 4C) ------------------------
__global__ void k_merge_gather(const float* __restrict__ x, float* __restrict__ out,
                               int B, int H, int C) {
  int Ho = H / 2;
  long long tot = (long long)B * Ho * Ho * 4 * C;
  long long i = (long long)blockIdx.x * 256 + threadIdx.x;
  if (i >= tot) return;
  int c4 = (int)(i % (4 * C));
  int c = c4 % C, g = c4 / C;          // g: 0=(0,0) 1=(1,0) 2=(0,1) 3=(1,1)
  long long r = i / (4 * C);
  int ow = (int)(r % Ho);
  int oh = (int)((r / Ho) % Ho);
  int b  = (int)(r / ((long long)Ho * Ho));
  int dh = g & 1, dw = g >> 1;
  out[i] = x[(((long long)b * H + 2 * oh + dh) * H + 2 * ow + dw) * C + c];
}

// ------------------- token mean + final FC ---------------------------------
__global__ void k_mean_tokens(const float* __restrict__ x, float* __restrict__ feats,
                              int B, int T, int C, int stride, int off) {
  long long i = (long long)blockIdx.x * 256 + threadIdx.x;
  if (i >= (long long)B * C) return;
  int c = (int)(i % C), b = (int)(i / C);
  float s = 0.f;
  for (int t = 0; t < T; ++t) s += x[((long long)b * T + t) * C + c];
  feats[(long long)b * stride + off + c] = s / T;
}

__global__ void k_final_fc(const float* __restrict__ feats, const float* __restrict__ csv,
                           const float* __restrict__ fcw, const float* __restrict__ fcb,
                           float* __restrict__ out) {
  // 64 blocks of one wave: block -> (b, o); K = 3*768 + 107 = 2411
  int lane = threadIdx.x;
  int b = blockIdx.x >> 1, o = blockIdx.x & 1;
  const float* wrow = fcw + o * 2411;
  float s = 0.f;
  for (int j = lane; j < 2411; j += 32) {
    float xv = (j < 2304) ? feats[b * 2304 + j] : csv[b * 107 + (j - 2304)];
    s += xv * wrow[j];
  }
  for (int off = 16; off; off >>= 1) s += __shfl_xor(s, off, 32);
  if (lane == 0) out[b * 2 + o] = s + fcb[o];
}

// ---------------------------------------------------------------------------

static inline int cdiv(long long a, long long b) { return (int)((a + b - 1) / b); }
static inline size_t aln(size_t x) { return (x + 255) & ~(size_t)255; }

extern "C" void kernel_launch(void* const* d_in, const int* in_sizes, int n_in,
                              void* d_out, int out_size, void* d_ws, size_t ws_size,
                              hipStream_t stream) {
  (void)in_sizes; (void)n_in; (void)out_size; (void)ws_size;
  const int B = 32;
  static const int DEPTHS_[4] = {2, 2, 6, 2};
  static const int HEADS_[4]  = {3, 6, 12, 24};

  // ---- parameter pointers (JAX pytree sorted-key order) ----
  auto F = [&](int i) { return (const float*)d_in[i]; };
  int idx = 4;
  const float* p_fc_b    = F(idx++);
  const float* p_fc_w    = F(idx++);
  const float* p_norm_b  = F(idx++);
  const float* p_norm_g  = F(idx++);
  const float* p_patch_b = F(idx++);
  const float* p_patch_w = F(idx++);
  struct Blk { const float *fc1_b,*fc1_w,*fc2_b,*fc2_w,*ln1_b,*ln1_g,*ln2_b,*ln2_g,
                            *proj_b,*proj_w,*qkv_b,*qkv_w,*rpb; };
  struct Mrg { const float *ln_b,*ln_g,*red_w; };
  Blk blocks[4][6]; Mrg merges[3];
  for (int si = 0; si < 4; ++si) {
    for (int bi = 0; bi < DEPTHS_[si]; ++bi) {
      Blk& bl = blocks[si][bi];
      bl.fc1_b = F(idx++); bl.fc1_w = F(idx++); bl.fc2_b = F(idx++); bl.fc2_w = F(idx++);
      bl.ln1_b = F(idx++); bl.ln1_g = F(idx++); bl.ln2_b = F(idx++); bl.ln2_g = F(idx++);
      bl.proj_b = F(idx++); bl.proj_w = F(idx++); bl.qkv_b = F(idx++); bl.qkv_w = F(idx++);
      bl.rpb = F(idx++);
    }
    if (si < 3) { merges[si].ln_b = F(idx++); merges[si].ln_g = F(idx++); merges[si].red_w = F(idx++); }
  }

  // ---- workspace arena (stage-0 maxima; ~430 MB total) ----
  char* p = (char*)d_ws;
  auto carve = [&](size_t bytes) { char* r = p; p += aln(bytes); return r; };
  float* X    = (float*)carve(9633792ull  * 4);   // activations (B*L, C)
  bf16*  XN   = (bf16*) carve(9633792ull  * 2);   // LN output bf16
  bf16*  WIN  = (bf16*) carve(12582912ull * 2);   // windowed xn / im2col
  float* QKV  = (float*)carve(37748736ull * 4);   // qkv GEMM out
  bf16*  ATT  = (bf16*) carve(12582912ull * 2);   // attention out bf16
  float* PROJ = (float*)carve(12582912ull * 4);   // proj / merge / final-LN out
  bf16*  FC1  = (bf16*) carve(38535168ull * 2);   // fc1(+gelu) out bf16
  float* FC2  = (float*)carve(9633792ull  * 4);   // fc2 out / merge-gather
  bf16*  WB   = (bf16*) carve(2359296ull  * 2);   // weight scratch bf16
  float* FEATS= (float*)carve(32ull * 2304 * 4);  // (B, 3*768)

  // ---- launch helpers ----
  auto conv = [&](const float* src, bf16* dst, int N, int K, int Kpad) {
    long long n = (long long)N * Kpad;
    k_convert_pad<<<cdiv(n, 256), 256, 0, stream>>>(src, dst, N, K, Kpad);
  };
  auto gemm = [&](const bf16* A, int lda, const bf16* W, int ldw, const float* bias,
                  void* out, int outBf16, int gelu, int M, int N, int K) {
    dim3 g(cdiv(M / 32, 4), cdiv(N, 64)), b(32, 4);
    k_gemm<<<g, b, 0, stream>>>(A, lda, W, ldw, bias, out, outBf16, gelu, M, N, K);
  };
  auto lnorm = [&](const float* x, const float* g, const float* b, void* out,
                   int outBf16, long long rows, int C) {
    k_layernorm<<<(unsigned)rows, 32, 0, stream>>>(x, g, b, out, outBf16, C);
  };

  // ---- three modalities: DCE, T2, T1 (concat order of the reference) ----
  for (int m = 0; m < 3; ++m) {
    const float* img = (const float*)d_in[m];

    // patch embed: conv(3->96, k4 s4) as im2col(K=48 pad 64) GEMM
    {
      long long tot = (long long)B * 3136 * 64;
      k_im2col<<<cdiv(tot, 256), 256, 0, stream>>>(img, WIN, B);
    }
    conv(p_patch_w, WB, 96, 48, 64);
    gemm(WIN, 64, WB, 64, p_patch_b, X, 0, 0, B * 3136, 96, 64);

    int C = 96, H = 56;
    for (int si = 0; si < 4; ++si) {
      int heads = HEADS_[si];
      int L = H * H, nWr = H / 7, nWimg = nWr * nWr;
      int rowsW = B * nWimg * 64;
      for (int bi = 0; bi < DEPTHS_[si]; ++bi) {
        const Blk& bl = blocks[si][bi];
        int shift = (bi % 2 == 1 && H > 7) ? 3 : 0;

        // --- window attention branch ---
        lnorm(X, bl.ln1_g, bl.ln1_b, XN, 1, (long long)B * L, C);
        {
          long long tot = (long long)rowsW * C;
          k_window_gather<<<cdiv(tot, 256), 256, 0, stream>>>(XN, WIN, B, H, C, shift);
        }
        conv(bl.qkv_w, WB, 3 * C, C, C);
        gemm(WIN, C, WB, C, bl.qkv_b, QKV, 0, 0, rowsW, 3 * C, C);
        k_window_attn<<<B * nWimg * heads, 256, 0, stream>>>(QKV, bl.rpb, ATT, C, heads,
                                                             H, shift, nWimg);
        conv(bl.proj_w, WB, C, C, C);
        gemm(ATT, C, WB, C, bl.proj_b, PROJ, 0, 0, rowsW, C, C);
        {
          long long tot = (long long)B * H * H * C;
          k_scatter_add<<<cdiv(tot, 256), 256, 0, stream>>>(PROJ, X, B, H, C, shift);
        }

        // --- MLP branch ---
        lnorm(X, bl.ln2_g, bl.ln2_b, XN, 1, (long long)B * L, C);
        conv(bl.fc1_w, WB, 4 * C, C, C);
        gemm(XN, C, WB, C, bl.fc1_b, FC1, 1, 1, B * L, 4 * C, C);   // +exact GELU
        conv(bl.fc2_w, WB, C, 4 * C, 4 * C);
        gemm(FC1, 4 * C, WB, 4 * C, bl.fc2_b, FC2, 0, 0, B * L, C, 4 * C);
        {
          long long n = (long long)B * L * C;
          k_add<<<cdiv(n, 256), 256, 0, stream>>>(X, FC2, n);
        }
      }
      if (si < 3) {
        long long rowsM = (long long)B * L / 4;
        {
          long long tot = rowsM * 4 * C;
          k_merge_gather<<<cdiv(tot, 256), 256, 0, stream>>>(X, FC2, B, H, C);
        }
        lnorm(FC2, merges[si].ln_g, merges[si].ln_b, XN, 1, rowsM, 4 * C);
        conv(merges[si].red_w, WB, 2 * C, 4 * C, 4 * C);
        gemm(XN, 4 * C, WB, 4 * C, nullptr, PROJ, 0, 0, (int)rowsM, 2 * C, 4 * C);
        hipMemcpyAsync(X, PROJ, (size_t)rowsM * 2 * C * sizeof(float),
                       hipMemcpyDeviceToDevice, stream);
        C *= 2; H /= 2;
      }
    }

    // final norm + token mean -> feats slice
    lnorm(X, p_norm_g, p_norm_b, PROJ, 0, (long long)B * 49, 768);
    {
      long long tot = (long long)B * 768;
      k_mean_tokens<<<cdiv(tot, 256), 256, 0, stream>>>(PROJ, FEATS, B, 49, 768,
                                                        2304, m * 768);
    }
  }

  k_final_fc<<<64, 32, 0, stream>>>(FEATS, (const float*)d_in[3], p_fc_w, p_fc_b,
                                    (float*)d_out);
}
